// MindForgeLoRAHead_35459249996096
// MI455X (gfx1250) — matrix operand order repaired and checked by reading, not compiled, not run on hardware
//
#include <hip/hip_runtime.h>
#include <hip/hip_bf16.h>

// ---------------- problem constants ----------------
#define D_MODEL   2048
#define C_CLASSES 50257
#define C_PAD     50304      // 393 * 128
#define N_BASIS   8
#define RANK      4
#define HIDDEN    128
#define B_ROWS    2048
#define LN_EPS    1e-5f

// ---------------- vector types ----------------
typedef __attribute__((ext_vector_type(16))) __bf16    bf16x16;
typedef __attribute__((ext_vector_type(8)))  __bf16    bf16x8;
typedef __attribute__((ext_vector_type(8)))  float     f32x8;
typedef __attribute__((ext_vector_type(4)))  float     f32x4;
typedef __attribute__((ext_vector_type(4)))  unsigned  ui32x4;
typedef __attribute__((ext_vector_type(8)))  int       i32x8;
typedef __attribute__((ext_vector_type(4)))  int       i32x4;

#if __has_builtin(__builtin_amdgcn_tensor_load_to_lds)
#define HAVE_TDM 1
#else
#define HAVE_TDM 0
#endif

// scalar fp32 -> bf16 (RTNE) for the tiny prep paths
static __device__ __forceinline__ __bf16 f2bf(float f) {
  unsigned u = __builtin_bit_cast(unsigned, f);
  u += 0x7fffu + ((u >> 16) & 1u);
  unsigned short h = (unsigned short)(u >> 16);
  return __builtin_bit_cast(__bf16, h);
}

static __device__ __forceinline__ bf16x16 cat8(bf16x8 a, bf16x8 b) {
  return __builtin_shufflevector(a, b, 0,1,2,3,4,5,6,7,8,9,10,11,12,13,14,15);
}
static __device__ __forceinline__ f32x8 cat4f(f32x4 a, f32x4 b) {
  return __builtin_shufflevector(a, b, 0,1,2,3,4,5,6,7);
}

// A-matrix fragment (16x32 bf16, MxK), ISA 7.12.2 lane layout
static __device__ __forceinline__ bf16x16 frag_a(const __bf16* base, int lane, int stride) {
  const int m = lane & 15;
  const int h = (lane >> 4) << 3;
  const bf16x8 lo = *(const bf16x8*)(base + (size_t)m * stride + h);
  const bf16x8 hi = *(const bf16x8*)(base + (size_t)m * stride + 16 + h);
  return cat8(lo, hi);
}

// B-matrix fragment (32x16 bf16, KxN), stored N-major (column contiguous in K)
static __device__ __forceinline__ bf16x16 frag_b(const __bf16* base, int lane, int stride) {
  const int n  = lane & 15;
  const int kh = (lane >> 4) << 3;
  const bf16x8 lo = *(const bf16x8*)(base + (size_t)n * stride + kh);
  const bf16x8 hi = *(const bf16x8*)(base + (size_t)n * stride + 16 + kh);
  return cat8(lo, hi);
}

#if HAVE_TDM
// ---- Tensor Data Mover: one 16-row x 32-elem bf16 tile per wave ----
// D# per ISA 8.3/8.4. pad fields re-create the 80B padded LDS rows:
// pad_interval=3 (64B) + pad_amount=3 (16B) -> LDS stride 80B.
static __device__ __forceinline__ void tdm_load_x_tile(const __bf16* gsrc, __bf16* ldst) {
  const unsigned long long ga = (unsigned long long)(uintptr_t)gsrc;
  ui32x4 g0;
  g0[0] = 1u;                                        // count=1 (valid user D#)
  g0[1] = (unsigned)(uintptr_t)ldst;                 // lds_addr (bytes)
  g0[2] = (unsigned)ga;                              // global_addr[31:0]
  g0[3] = ((unsigned)(ga >> 32) & 0x01FFFFFFu) | 0x80000000u;  // addr[56:32] | type=2
  i32x8 g1;
  g1[0] = 0x06D10000;        // data_size=2B | pad_enable | pad_interval=64B | pad_amount=16B
  g1[1] = (int)(D_MODEL << 16);   // tensor_dim0 = 2048 (low16 at [63:48])
  g1[2] = (int)(B_ROWS  << 16);   // tensor_dim0 hi=0 ; tensor_dim1 = 2048 (low16)
  g1[3] = (int)(32 << 16);        // tensor_dim1 hi=0 ; tile_dim0 = 32
  g1[4] = 16;                     // tile_dim1 = 16 rows ; tile_dim2 = 0
  g1[5] = D_MODEL;                // tensor_dim0_stride = 2048
  g1[6] = 0;
  g1[7] = 0;
  i32x4 z4 = {};                  // 2-D tile: groups 2/3 unused
  i32x8 z8 = {};                  // extra SGPR group (6-arg toolchain form): unused
  __builtin_amdgcn_tensor_load_to_lds(g0, g1, z4, z4, z8, 0);
}
static __device__ __forceinline__ void wait_x_tiles() {
  __builtin_amdgcn_s_wait_tensorcnt(0);
}
#else
// ---- fallback: per-lane async DMA to LDS (ASYNCcnt) ----
static __device__ __forceinline__ void async_b128(const __bf16* g, __bf16* lds) {
  unsigned l = (unsigned)(uintptr_t)lds;
  unsigned long long ga = (unsigned long long)(uintptr_t)g;
  asm volatile("global_load_async_to_lds_b128 %0, %1, off"
               :: "v"(l), "v"(ga) : "memory");
}
static __device__ __forceinline__ void wait_x_tiles() {
  asm volatile("s_wait_asynccnt 0x0" ::: "memory");
}
#endif

// =====================================================================
// Kernel 1: context MLP -> LN -> exact GELU -> coeffs -> z -> u (bf16)
// =====================================================================
__global__ __launch_bounds__(HIDDEN) void prep_kernel(
    const float* __restrict__ x, const float* __restrict__ context,
    const float* __restrict__ ctx_w, const float* __restrict__ ctx_b,
    const float* __restrict__ ln_g, const float* __restrict__ ln_b,
    const float* __restrict__ coeff_w, const float* __restrict__ coeff_b,
    const float* __restrict__ basis_A, __bf16* __restrict__ u_out)
{
  __shared__ __align__(16) float s_row[D_MODEL];
  __shared__ float s_h[HIDDEN];
  __shared__ float s_red[HIDDEN];
  __shared__ float s_coeff[N_BASIS];
  __shared__ float s_t[N_BASIS * RANK];
  __shared__ float s_z[RANK];

  const int b   = blockIdx.x;
  const int tid = threadIdx.x;

  for (int i = tid; i < D_MODEL; i += HIDDEN)
    s_row[i] = context[(size_t)b * D_MODEL + i];
  __syncthreads();

  float acc = 0.f;
  {
    const f32x4* wr = (const f32x4*)(ctx_w + (size_t)tid * D_MODEL);
    #pragma unroll 4
    for (int k4 = 0; k4 < D_MODEL / 4; ++k4) {
      f32x4 wv = wr[k4];
      f32x4 sv = *(const f32x4*)&s_row[k4 * 4];
      acc += wv[0]*sv[0] + wv[1]*sv[1] + wv[2]*sv[2] + wv[3]*sv[3];
    }
  }
  const float h0 = acc + ctx_b[tid];

  s_red[tid] = h0; __syncthreads();
  for (int s = HIDDEN / 2; s > 0; s >>= 1) {
    if (tid < s) s_red[tid] += s_red[tid + s];
    __syncthreads();
  }
  const float mu = s_red[0] * (1.f / HIDDEN);
  __syncthreads();
  const float dd = h0 - mu;
  s_red[tid] = dd * dd; __syncthreads();
  for (int s = HIDDEN / 2; s > 0; s >>= 1) {
    if (tid < s) s_red[tid] += s_red[tid + s];
    __syncthreads();
  }
  const float var = s_red[0] * (1.f / HIDDEN);

  const float hn = dd * rsqrtf(var + LN_EPS) * ln_g[tid] + ln_b[tid];
  const float g  = 0.5f * hn * (1.f + erff(hn * 0.70710678118654752f));
  s_h[tid] = g;
  __syncthreads();

  if (tid < N_BASIS) {
    float c = 0.f;
    #pragma unroll 4
    for (int k = 0; k < HIDDEN; ++k) c += s_h[k] * coeff_w[tid * HIDDEN + k];
    s_coeff[tid] = c + coeff_b[tid];
  }
  __syncthreads();

  for (int i = tid; i < D_MODEL; i += HIDDEN)
    s_row[i] = x[(size_t)b * D_MODEL + i];
  __syncthreads();

  {
    const int nr   = tid & 31;
    const int part = tid >> 5;
    const f32x4* ar = (const f32x4*)(basis_A + (size_t)nr * D_MODEL + part * 512);
    const f32x4* xr = (const f32x4*)&s_row[part * 512];
    float p = 0.f;
    #pragma unroll 4
    for (int k4 = 0; k4 < 128; ++k4) {
      f32x4 av = ar[k4];
      f32x4 xv = xr[k4];
      p += av[0]*xv[0] + av[1]*xv[1] + av[2]*xv[2] + av[3]*xv[3];
    }
    s_red[tid] = p;
  }
  __syncthreads();
  if (tid < N_BASIS * RANK)
    s_t[tid] = s_red[tid] + s_red[tid + 32] + s_red[tid + 64] + s_red[tid + 96];
  __syncthreads();

  if (tid < RANK) {
    float z = 0.f;
    #pragma unroll
    for (int n = 0; n < N_BASIS; ++n) z += s_coeff[n] * s_t[n * RANK + tid];
    s_z[tid] = z;
  }
  __syncthreads();

  if (tid < N_BASIS * RANK) {
    const int n = tid >> 2, r = tid & 3;
    u_out[(size_t)b * 32 + tid] = f2bf(s_coeff[n] * s_z[r]);
  }
}

// =====================================================================
// Kernel 2: BmatT[c][k] = basis_B[k/4, c, k%4] in bf16, padded to C_PAD
// =====================================================================
__global__ __launch_bounds__(256) void bmat_kernel(
    const float* __restrict__ basis_B, __bf16* __restrict__ bmatT)
{
  const int c = blockIdx.x * 256 + threadIdx.x;
  if (c >= C_PAD) return;
  #pragma unroll
  for (int nr = 0; nr < N_BASIS * RANK; ++nr) {
    const int n = nr >> 2, r = nr & 3;
    float v = 0.f;
    if (c < C_CLASSES) v = basis_B[((size_t)n * C_CLASSES + c) * RANK + r];
    bmatT[(size_t)c * 32 + nr] = f2bf(v);
  }
}

// =====================================================================
// Kernel 3: pre-convert x to bf16 once
// =====================================================================
__global__ __launch_bounds__(256) void xcvt_kernel(
    const float* __restrict__ x, __bf16* __restrict__ xbf)
{
  const size_t i = ((size_t)blockIdx.x * 256 + threadIdx.x) * 8;
  f32x4 a = *(const f32x4*)(x + i);
  f32x4 b = *(const f32x4*)(x + i + 4);
  *(bf16x8*)(xbf + i) = __builtin_convertvector(cat4f(a, b), bf16x8);
}

// =====================================================================
// Kernel 4: out = x @ base_w^T + base_b + u @ Bmat
// 393 blocks x all rows: base_w streamed from HBM once. Double-buffered
// LDS; x tile staged by the Tensor Data Mover (one tensor_load_to_lds
// per wave, TENSORcnt), w tile via regs + v_cvt_pk_bf16_f32.
// =====================================================================
__global__ __launch_bounds__(256, 1) void gemm_kernel(
    const __bf16* __restrict__ xbf, const float* __restrict__ base_w,
    const float* __restrict__ base_b, const __bf16* __restrict__ u,
    const __bf16* __restrict__ bmatT, float* __restrict__ out)
{
  constexpr int KC      = 32;
  constexpr int NK      = D_MODEL / KC;   // 64
  constexpr int LDS_STR = 40;             // 80B rows (TDM pad reproduces this)

  __shared__ __align__(16) __bf16 sA[2][128 * LDS_STR];
  __shared__ __align__(16) __bf16 sW[2][128 * LDS_STR];

  const int tid  = threadIdx.x;
  const int lane = tid & 31;
  const int wave = tid >> 5;
  const int wvu  = __builtin_amdgcn_readfirstlane(wave);  // wave-uniform id for TDM
  const int wm   = wave >> 1;
  const int wn   = wave & 1;
  const int nb   = blockIdx.x * 128;

  // per-thread staging coordinates for the w tile (512 16B chunks / 128x32)
  const int r0 = tid >> 2,         c0 = tid & 3;
  const int r1 = (tid + 256) >> 2, c1 = tid & 3;

  for (int mt = 0; mt < B_ROWS / 128; ++mt) {
    const int mb = mt * 128;
    f32x8 acc[2][4] = {};

    // ---- prologue: stage k-chunk 0 into buffer 0 ----
#if HAVE_TDM
    tdm_load_x_tile(xbf + (size_t)(mb + wvu * 16) * D_MODEL,
                    &sA[0][(wvu * 16) * LDS_STR]);
#else
    async_b128(xbf + (size_t)(mb + r0) * D_MODEL + c0 * 8, &sA[0][r0 * LDS_STR + c0 * 8]);
    async_b128(xbf + (size_t)(mb + r1) * D_MODEL + c1 * 8, &sA[0][r1 * LDS_STR + c1 * 8]);
#endif
    {
      #pragma unroll
      for (int i = 0; i < 2; ++i) {
        const int row = i ? r1 : r0, ch = i ? c1 : c0;
        f32x4 a = {}, b = {};
        const int c = nb + row;
        if (c < C_CLASSES) {
          const float* p = base_w + (size_t)c * D_MODEL + ch * 8;
          a = *(const f32x4*)p; b = *(const f32x4*)(p + 4);
        }
        *(bf16x8*)&sW[0][row * LDS_STR + ch * 8] = __builtin_convertvector(cat4f(a, b), bf16x8);
      }
    }
    wait_x_tiles();
    __syncthreads();

    // ---- main K loop, double buffered, one barrier per iteration ----
    for (int ki = 0; ki < NK; ++ki) {
      const int cur = ki & 1, nxt = cur ^ 1;
      const int kcn = (ki + 1) * KC;
      const bool pf = (ki + 1 < NK);

      f32x4 wr0a = {}, wr0b = {}, wr1a = {}, wr1b = {};
      if (pf) {
#if HAVE_TDM
        tdm_load_x_tile(xbf + (size_t)(mb + wvu * 16) * D_MODEL + kcn,
                        &sA[nxt][(wvu * 16) * LDS_STR]);
#else
        async_b128(xbf + (size_t)(mb + r0) * D_MODEL + kcn + c0 * 8, &sA[nxt][r0 * LDS_STR + c0 * 8]);
        async_b128(xbf + (size_t)(mb + r1) * D_MODEL + kcn + c1 * 8, &sA[nxt][r1 * LDS_STR + c1 * 8]);
#endif
        if (nb + r0 < C_CLASSES) {
          const float* p = base_w + (size_t)(nb + r0) * D_MODEL + kcn + c0 * 8;
          wr0a = *(const f32x4*)p; wr0b = *(const f32x4*)(p + 4);
        }
        if (nb + r1 < C_CLASSES) {
          const float* p = base_w + (size_t)(nb + r1) * D_MODEL + kcn + c1 * 8;
          wr1a = *(const f32x4*)p; wr1b = *(const f32x4*)(p + 4);
        }
      }

      // ---- fragments + 2x4 WMMA on current buffer ----
      bf16x16 af[2], bfr[4];
      #pragma unroll
      for (int i = 0; i < 2; ++i)
        af[i] = frag_a(&sA[cur][(wm * 32 + i * 16) * LDS_STR], lane, LDS_STR);
      #pragma unroll
      for (int j = 0; j < 4; ++j)
        bfr[j] = frag_b(&sW[cur][(wn * 64 + j * 16) * LDS_STR], lane, LDS_STR);

      #pragma unroll
      for (int i = 0; i < 2; ++i)
        #pragma unroll
        for (int j = 0; j < 4; ++j)
          acc[i][j] = __builtin_amdgcn_wmma_f32_16x16x32_bf16(
              false, af[i], false, bfr[j], (short)0, acc[i][j], false, false);

      if (pf) {
        *(bf16x8*)&sW[nxt][r0 * LDS_STR + c0 * 8] = __builtin_convertvector(cat4f(wr0a, wr0b), bf16x8);
        *(bf16x8*)&sW[nxt][r1 * LDS_STR + c1 * 8] = __builtin_convertvector(cat4f(wr1a, wr1b), bf16x8);
        wait_x_tiles();
      }
      __syncthreads();
    }

    // ---- epilogue: LoRA delta via one extra WMMA + bias + store ----
    bf16x16 ua[2];
    #pragma unroll
    for (int i = 0; i < 2; ++i)
      ua[i] = frag_a(u + (size_t)(mb + wm * 32 + i * 16) * 32, lane, 32);

    #pragma unroll
    for (int j = 0; j < 4; ++j) {
      const int cb = nb + wn * 64 + j * 16;
      bf16x16 bb = frag_b(bmatT + (size_t)cb * 32, lane, 32);
      #pragma unroll
      for (int i = 0; i < 2; ++i)
        acc[i][j] = __builtin_amdgcn_wmma_f32_16x16x32_bf16(
            false, ua[i], false, bb, (short)0, acc[i][j], false, false);
    }

    #pragma unroll
    for (int j = 0; j < 4; ++j) {
      const int c = nb + wn * 64 + j * 16 + (lane & 15);
      const float bias = (c < C_CLASSES) ? base_b[c] : 0.f;
      #pragma unroll
      for (int i = 0; i < 2; ++i) {
        f32x8 r = acc[i][j] + bias;
        const int mrow = mb + wm * 32 + i * 16 + ((lane >> 4) << 3);
        if (c < C_CLASSES) {
          #pragma unroll
          for (int e = 0; e < 8; ++e)
            out[(size_t)(mrow + e) * C_CLASSES + c] = r[e];
        }
      }
    }
  }
}

// =====================================================================
extern "C" void kernel_launch(void* const* d_in, const int* in_sizes, int n_in,
                              void* d_out, int out_size, void* d_ws, size_t ws_size,
                              hipStream_t stream) {
  const float* x       = (const float*)d_in[0];
  const float* context = (const float*)d_in[1];
  const float* base_w  = (const float*)d_in[2];
  const float* base_b  = (const float*)d_in[3];
  const float* ctx_w   = (const float*)d_in[4];
  const float* ctx_b   = (const float*)d_in[5];
  const float* ln_g    = (const float*)d_in[6];
  const float* ln_b    = (const float*)d_in[7];
  const float* coeff_w = (const float*)d_in[8];
  const float* coeff_b = (const float*)d_in[9];
  const float* basis_A = (const float*)d_in[10];
  const float* basis_B = (const float*)d_in[11];
  float* out = (float*)d_out;

  // workspace: [BmatT: C_PAD*32 bf16][u: B*32 bf16][xbf: B*D bf16]  (~11.7 MB)
  __bf16* bmatT = (__bf16*)d_ws;
  __bf16* u     = bmatT + (size_t)C_PAD * 32;
  __bf16* xbf   = u + (size_t)B_ROWS * 32;

  prep_kernel<<<B_ROWS, HIDDEN, 0, stream>>>(x, context, ctx_w, ctx_b, ln_g, ln_b,
                                             coeff_w, coeff_b, basis_A, u);
  bmat_kernel<<<C_PAD / 256, 256, 0, stream>>>(basis_B, bmatT);
  xcvt_kernel<<<(B_ROWS * D_MODEL) / (256 * 8), 256, 0, stream>>>(x, xbf);
  gemm_kernel<<<C_PAD / 128, 256, 0, stream>>>(xbf, base_w, base_b, u, bmatT, out);
}
